// GatingNetwork_36575941492950
// MI455X (gfx1250) — matrix operand order repaired and checked by reading, not compiled, not run on hardware
//
#include <hip/hip_runtime.h>

#define D_IN 768
#define H1   256
#define H2   128
#define NE   4
#define ROWS 32      // rows per workgroup
#define KCH  128     // K-chunk for stage 1 (6 chunks)
#define XBP  136     // LDS pitch (bf16) for 128-wide tiles: 272B rows, 16B aligned
#define H1P  264     // LDS pitch (bf16) for h1 (256-wide): 528B rows
#define H2P  136     // LDS pitch (bf16) for h2

typedef __bf16 bf16_t;
typedef __attribute__((ext_vector_type(16))) __bf16        v16bf;
typedef __attribute__((ext_vector_type(8)))  float         v8f;
typedef __attribute__((ext_vector_type(4)))  unsigned int  v4u;
typedef __attribute__((ext_vector_type(8)))  unsigned int  v8u;

static __device__ __forceinline__ unsigned pack_bf16(float lo, float hi) {
    unsigned short l = __builtin_bit_cast(unsigned short, (__bf16)lo);
    unsigned short h = __builtin_bit_cast(unsigned short, (__bf16)hi);
    return (unsigned)l | ((unsigned)h << 16);   // lowers to v_cvt_pk_bf16_f32
}

static __device__ __forceinline__ v16bf make_frag(v4u lo, v4u hi) {
    v8u u;
    u[0]=lo[0]; u[1]=lo[1]; u[2]=lo[2]; u[3]=lo[3];
    u[4]=hi[0]; u[5]=hi[1]; u[6]=hi[2]; u[7]=hi[3];
    return __builtin_bit_cast(v16bf, u);
}

// A fragment (16x32) from bf16 LDS tile.
// ISA layout: lanes 0-15 hold M=lane, K {0..7,16..23}; lanes 16-31 hold K {8..15,24..31}.
static __device__ __forceinline__ v16bf load_a_lds(const bf16_t* base, int pitch, int m0, int k0) {
    const int lane = threadIdx.x & 31;
    const bf16_t* p = base + (m0 + (lane & 15)) * pitch + k0 + ((lane >> 4) << 3);
    v4u lo = *reinterpret_cast<const v4u*>(p);        // K +0..7   (VGPR 0-3)
    v4u hi = *reinterpret_cast<const v4u*>(p + 16);   // K +16..23 (VGPR 4-7)
    return make_frag(lo, hi);
}

// B fragment (32x16) from global W^T [N][K] bf16.
// ISA layout: lanes 0-15 hold N=lane, K {0..7,16..23}; lanes 16-31 hold K {8..15,24..31}.
static __device__ __forceinline__ v16bf load_b_glb(const bf16_t* Wt, int pitchK, int n0, int k0) {
    const int lane = threadIdx.x & 31;
    const bf16_t* p = Wt + (size_t)(n0 + (lane & 15)) * pitchK + k0 + ((lane >> 4) << 3);
    v4u lo = *reinterpret_cast<const v4u*>(p);        // K +0..7
    v4u hi = *reinterpret_cast<const v4u*>(p + 16);   // K +16..23
    return make_frag(lo, hi);
}

// ---- prep: transpose + convert W1, W2 to bf16 (kept L2-resident, reused by all WGs) ----
__global__ void __launch_bounds__(256)
prep_weights(const float* __restrict__ W1, const float* __restrict__ W2,
             bf16_t* __restrict__ W1t, bf16_t* __restrict__ W2t) {
    int i = blockIdx.x * blockDim.x + threadIdx.x;
    if (i < H1 * D_IN) {                       // W1t[n][k] = W1[k][n]
        int n = i / D_IN, k = i % D_IN;
        W1t[i] = (bf16_t)W1[k * H1 + n];
    } else {
        int j = i - H1 * D_IN;
        if (j < H2 * H1) {                     // W2t[n][k] = W2[k][n]
            int n = j / H1, k = j % H1;
            W2t[j] = (bf16_t)W2[k * H2 + n];
        }
    }
}

// ---- fused router: x -> h1 -> h2 -> logits -> top2 softmax gates ----
__global__ void __launch_bounds__(256)
router_kernel(const float* __restrict__ x,
              const bf16_t* __restrict__ W1t, const float* __restrict__ b1,
              const bf16_t* __restrict__ W2t, const float* __restrict__ b2,
              const float* __restrict__ W3,  const float* __restrict__ b3,
              float* __restrict__ gates, int* __restrict__ idxout) {
    __shared__ __align__(16) float  xsf[ROWS * KCH];  // x chunk f32 (async DMA target), 16KB
    __shared__ __align__(16) bf16_t xsb[ROWS * XBP];  // x chunk bf16 (WMMA A source)
    __shared__ __align__(16) bf16_t h1s[ROWS * H1P];  // h1 (32x256) bf16
    __shared__ __align__(16) bf16_t h2s[ROWS * H2P];  // h2 (32x128) bf16
    __shared__ float ls[ROWS * NE];                   // logits

    const int tid  = threadIdx.x;
    const int lane = tid & 31;
    const int w    = tid >> 5;       // 8 waves
    const int wm   = w >> 2;         // 0..1 : 16-row slab
    const int wn   = w & 3;          // 0..3 : N slab
    const int row0 = blockIdx.x * ROWS;
    const int m0   = wm * 16;

    v8f zero;
    #pragma unroll
    for (int j = 0; j < 8; ++j) zero[j] = 0.0f;

    // ---------------- stage 1: h1 = relu(x @ W1 + b1), accumulate over 6 K-chunks of 128
    v8f acc[4] = {zero, zero, zero, zero};
    for (int kc = 0; kc < D_IN / KCH; ++kc) {
        // async-DMA x chunk [32 x 128] f32 into LDS: 1024 b128 units, 4 per thread, coalesced
        #pragma unroll
        for (int it = 0; it < 4; ++it) {
            const int u = tid + it * 256;
            const int r = u >> 5, c = u & 31;          // 32 units per row
            const float*   gp  = x + (size_t)(row0 + r) * D_IN + kc * KCH + (c << 2);
            const unsigned lof = (unsigned)(size_t)(&xsf[r * KCH + (c << 2)]);
            asm volatile("global_load_async_to_lds_b128 %0, %1, off"
                         :: "v"(lof), "v"(gp) : "memory");
        }
        if (kc + 1 < D_IN / KCH) {   // hint next chunk toward L2 (global_prefetch_b8)
            int r = tid >> 3, c = (tid & 7) << 4;
            __builtin_prefetch(&x[(size_t)(row0 + r) * D_IN + (kc + 1) * KCH + c], 0, 0);
        }
        asm volatile("s_wait_asynccnt 0x0" ::: "memory");
        __syncthreads();

        // one-shot cooperative convert: 2048 f32 pairs -> bf16 (8 per thread, conflict-free)
        #pragma unroll
        for (int it = 0; it < 8; ++it) {
            const int u = tid + it * 256;              // pair index
            const int r = u >> 6, cp = u & 63;         // 64 pairs per row
            const float* p = &xsf[r * KCH + (cp << 1)];
            *reinterpret_cast<unsigned*>(&xsb[r * XBP + (cp << 1)]) = pack_bf16(p[0], p[1]);
        }
        __syncthreads();

        #pragma unroll
        for (int ks = 0; ks < 4; ++ks) {
            const int k  = ks * 32;
            const int kg = kc * KCH + k;
            v16bf a = load_a_lds(xsb, XBP, m0, k);
            #pragma unroll
            for (int t = 0; t < 4; ++t) {
                v16bf b = load_b_glb(W1t, D_IN, wn * 64 + t * 16, kg);
                acc[t] = __builtin_amdgcn_wmma_f32_16x16x32_bf16(
                    false, a, false, b, (short)0, acc[t], false, false);
            }
        }
        __syncthreads();   // protect xsf/xsb before next chunk overwrites
    }
    // epilogue: bias + relu -> h1s (C layout: VGPR j -> M = j + 8*(lane>=16), N = lane%16)
    {
        const int mo = (lane >> 4) << 3;
        #pragma unroll
        for (int t = 0; t < 4; ++t) {
            const int n = wn * 64 + t * 16 + (lane & 15);
            const float bias = b1[n];
            #pragma unroll
            for (int j = 0; j < 8; ++j) {
                float v = acc[t][j] + bias;
                h1s[(m0 + mo + j) * H1P + n] = (bf16_t)(v > 0.0f ? v : 0.0f);
            }
        }
    }
    __syncthreads();

    // ---------------- stage 2: h2 = relu(h1 @ W2 + b2), K=256 all in LDS
    {
        v8f acc2[2] = {zero, zero};
        const int n0 = wn * 32;
        #pragma unroll
        for (int ks = 0; ks < 8; ++ks) {
            const int k = ks * 32;
            v16bf a = load_a_lds(h1s, H1P, m0, k);
            #pragma unroll
            for (int t = 0; t < 2; ++t) {
                v16bf b = load_b_glb(W2t, H1, n0 + t * 16, k);
                acc2[t] = __builtin_amdgcn_wmma_f32_16x16x32_bf16(
                    false, a, false, b, (short)0, acc2[t], false, false);
            }
        }
        const int mo = (lane >> 4) << 3;
        #pragma unroll
        for (int t = 0; t < 2; ++t) {
            const int n = n0 + t * 16 + (lane & 15);
            const float bias = b2[n];
            #pragma unroll
            for (int j = 0; j < 8; ++j) {
                float v = acc2[t][j] + bias;
                h2s[(m0 + mo + j) * H2P + n] = (bf16_t)(v > 0.0f ? v : 0.0f);
            }
        }
    }
    __syncthreads();

    // ---------------- stage 3: logits (K=128, N=4) — VALU dot products
    if (tid < ROWS * NE) {
        const int r = tid >> 2, e = tid & 3;
        float s = b3[e];
        #pragma unroll 8
        for (int k = 0; k < H2; ++k)
            s += (float)h2s[r * H2P + k] * W3[k * NE + e];
        ls[r * NE + e] = s;   // TEMPERATURE == 1.0
    }
    __syncthreads();

    // ---------------- top-2 of 4, softmax over the pair, scatter gates
    if (tid < ROWS) {
        const int r = tid;
        float lv[4];
        #pragma unroll
        for (int j = 0; j < 4; ++j) lv[j] = ls[r * NE + j];
        int i0 = 0; float v0 = lv[0];
        #pragma unroll
        for (int j = 1; j < 4; ++j) if (lv[j] > v0) { v0 = lv[j]; i0 = j; }   // first max (jax tie rule)
        int i1 = -1; float v1 = -3.4e38f;
        #pragma unroll
        for (int j = 0; j < 4; ++j) {
            if (j == i0) continue;
            if (lv[j] > v1) { v1 = lv[j]; i1 = j; }
        }
        const float e1 = __expf(v1 - v0);     // v1 <= v0, numerically safe
        const float g0 = 1.0f / (1.0f + e1);
        const float g1 = e1 / (1.0f + e1);
        float g[4] = {0.f, 0.f, 0.f, 0.f};
        g[i0] = g0; g[i1] = g1;
        const size_t ro = (size_t)(row0 + r);
        #pragma unroll
        for (int j = 0; j < 4; ++j) gates[ro * 4 + j] = g[j];
        idxout[ro * 2 + 0] = i0;
        idxout[ro * 2 + 1] = i1;
    }
}

extern "C" void kernel_launch(void* const* d_in, const int* in_sizes, int n_in,
                              void* d_out, int out_size, void* d_ws, size_t ws_size,
                              hipStream_t stream) {
    const float* x  = (const float*)d_in[0];
    const float* W1 = (const float*)d_in[1];
    const float* b1 = (const float*)d_in[2];
    const float* W2 = (const float*)d_in[3];
    const float* b2 = (const float*)d_in[4];
    const float* W3 = (const float*)d_in[5];
    const float* b3 = (const float*)d_in[6];

    const int Btot = in_sizes[0] / D_IN;          // 131072

    bf16_t* W1t = (bf16_t*)d_ws;                                  // [256][768] bf16
    bf16_t* W2t = (bf16_t*)((char*)d_ws + (size_t)H1 * D_IN * 2); // [128][256] bf16

    float* gates = (float*)d_out;                       // [B][4] f32
    int*   idxo  = (int*)(gates + (size_t)Btot * NE);   // [B][2] i32

    const int prepN = H1 * D_IN + H2 * H1;              // 229376
    prep_weights<<<(prepN + 255) / 256, 256, 0, stream>>>(W1, W2, W1t, W2t);
    router_kernel<<<Btot / ROWS, 256, 0, stream>>>(x, W1t, b1, W2t, b2, W3, b3, gates, idxo);
}
